// ScalarMPNN_17162689315166
// MI455X (gfx1250) — matrix operand order repaired and checked by reading, not compiled, not run on hardware
//
#include <hip/hip_runtime.h>

#define HID 64
#define NRBF 20
#define CUTOFF 4.0f
#define RBF_WIDTH ((CUTOFF / NRBF) * (CUTOFF / NRBF))   // 0.04
#define RBF_SPACING (CUTOFF / (NRBF - 1))
#define PI_F 3.14159265358979f

typedef __attribute__((ext_vector_type(16))) _Float16 v16h;
typedef __attribute__((ext_vector_type(8)))  float    v8f;

// Fast activations: v_exp_f32 + v_rcp_f32, no IEEE divide expansion.
static __device__ __forceinline__ float fast_sigmoid(float x) {
  return __builtin_amdgcn_rcpf(1.f + __expf(-x));
}
static __device__ __forceinline__ float fast_silu(float x) { return x * fast_sigmoid(x); }

// Guaranteed hardware float atomic (CDNA5 GLOBAL_ATOMIC_ADD_F32), no CAS fallback.
static __device__ __forceinline__ void atomic_add_f32_global(float* p, float v) {
  asm volatile("global_atomic_add_f32 %0, %1, off" :: "v"(p), "v"(v) : "memory");
}

// ---------------------------------------------------------------------------
// Fragment-contiguous LDS layouts: each lane's 16 halves are packed into one
// aligned 32-byte chunk -> fragment load = single v16h LDS read (2x ds_load_b128).
//
// B fragment (32x16): element (K,N): frag=(K>>5)*4+(N>>4), lane=((K>>3)&1)*16+(N&15), slot=((K>>1)&8)+(K&7).
// A fragment (16x32): element (M,rel): lane=M+16*((rel>>3)&1), slot=((rel>>1)&8)+(rel&7).
// ---------------------------------------------------------------------------
static __device__ __forceinline__ int bfrag_index(int K, int N) {
  return ((((K >> 5) << 2) + (N >> 4)) << 9) + (((((K >> 3) & 1) << 4) + (N & 15)) << 4) + (((K >> 1) & 8) + (K & 7));
}
static __device__ __forceinline__ int afrag_index(int kc, int M, int rel) {
  int g = (rel >> 3) & 1;
  int slot = ((rel >> 1) & 8) + (rel & 7);
  return (kc << 9) + ((M + (g << 4)) << 4) + slot;
}
static __device__ __forceinline__ v16h frag_load(const _Float16* base, int frag) {
  int lane = threadIdx.x & 31;
  return *(const v16h*)(base + (frag << 9) + (lane << 4));
}

#define WMMA_F16(a, b, c) \
  __builtin_amdgcn_wmma_f32_16x16x32_f16(false, (a), false, (b), (short)0, (c), false, false)

// ---------------------------------------------------------------------------
// h0 = silu(charges[:,None] @ embed_w + embed_b)
// ---------------------------------------------------------------------------
__global__ __launch_bounds__(256) void embed_kernel(const float* __restrict__ charges,
                                                    const float* __restrict__ w,
                                                    const float* __restrict__ b,
                                                    float* __restrict__ h, int total) {
  int i = blockIdx.x * 256 + threadIdx.x;
  if (i >= total) return;
  int c = i & (HID - 1);
  int node = i >> 6;
  float x = charges[node] * w[c] + b[c];
  h[i] = fast_silu(x);
}

// ---------------------------------------------------------------------------
// Fused edge pipeline: RBF -> GEMM(20x64) -> SiLU -> GEMM(64x64) -> sigmoid ->
// gate * h[src] -> atomic scatter-add into aggr[dst].
// One wave handles a 16-edge tile; 8 waves per block.
// ---------------------------------------------------------------------------
__global__ __launch_bounds__(256) void edge_kernel(
    const float* __restrict__ dist, const int* __restrict__ src, const int* __restrict__ dst,
    const float* __restrict__ w1, const float* __restrict__ b1,
    const float* __restrict__ w2, const float* __restrict__ b2,
    const float* __restrict__ h, float* __restrict__ aggr, int E) {
  __shared__ __align__(32) _Float16 w1h[4 * 512];       // 4 B-frags (K padded 20->32)
  __shared__ __align__(32) _Float16 w2h[8 * 512];       // 8 B-frags
  __shared__ __align__(32) _Float16 t_f[8][2 * 512];    // SiLU intermediate, A-frag layout
  __shared__ int sidx[8][16];
  __shared__ int didx[8][16];

  const int tid = threadIdx.x;
  const int wid = tid >> 5;
  const int lane = tid & 31;
  const int m = lane & 15;
  const int grp = lane >> 4;
  const int n = m;

  // Convert weights fp32 -> f16 directly into fragment-contiguous layout.
  for (int i = tid; i < 32 * HID; i += 256) {
    int K = i >> 6, N = i & (HID - 1);
    w1h[bfrag_index(K, N)] = (K < NRBF) ? (_Float16)w1[K * HID + N] : (_Float16)0.f;
  }
  for (int i = tid; i < 64 * HID; i += 256) {
    int K = i >> 6, N = i & (HID - 1);
    w2h[bfrag_index(K, N)] = (_Float16)w2[i];
  }
  __syncthreads();

  // Weight fragments held in registers for the whole kernel.
  v16h bw1[4];
#pragma unroll
  for (int nt = 0; nt < 4; ++nt) bw1[nt] = frag_load(w1h, nt);
  v16h bw2[8];
#pragma unroll
  for (int f = 0; f < 8; ++f) bw2[f] = frag_load(w2h, f);

  float b1v[4], b2v[4];
#pragma unroll
  for (int nt = 0; nt < 4; ++nt) {
    b1v[nt] = b1[nt * 16 + n];
    b2v[nt] = b2[nt * 16 + n];
  }

  const int ntiles = (E + 15) >> 4;
  const int waves_total = gridDim.x * 8;
  const int iters = (ntiles + waves_total - 1) / waves_total;

  for (int it = 0; it < iters; ++it) {
    const int tile = it * waves_total + blockIdx.x * 8 + wid;
    const bool active = (tile < ntiles);
    const long ebase = (long)tile << 4;

    // Build GEMM1 A-fragment directly in registers from the RBF expansion.
    float d = 0.f;
    bool ev = false;
    if (active) {
      long e = ebase + m;
      if (e < (long)E) { d = dist[e]; ev = true; }
    }
    float env = 0.5f * (1.f + __cosf(PI_F * d / CUTOFF));
    v16h a;
#pragma unroll
    for (int v = 0; v < 8; ++v) {
#pragma unroll
      for (int p = 0; p < 2; ++p) {
        int K = ((v & 4) ? 16 : 0) + grp * 8 + (v & 3) * 2 + p;
        float val = 0.f;
        if (ev && K < NRBF) {
          float t = d - (float)K * RBF_SPACING;
          val = __expf(-t * t / RBF_WIDTH) * env;
        }
        a[2 * v + p] = (_Float16)val;
      }
    }
    if (active && lane < 16) {
      long e = ebase + lane;
      sidx[wid][lane] = (e < (long)E) ? src[e] : 0;
      didx[wid][lane] = (e < (long)E) ? dst[e] : 0;
    }

    // GEMM1: rbf[16x32] x W1[32x64]  (WMMA outside divergent guards)
    v8f c1[4];
#pragma unroll
    for (int nt = 0; nt < 4; ++nt) {
      v8f z = {};
      c1[nt] = WMMA_F16(a, bw1[nt], z);
    }
    if (active) {
#pragma unroll
      for (int nt = 0; nt < 4; ++nt) {
        int kc = nt >> 1;
        int rel = ((nt & 1) << 4) + n;       // per-lane constant
#pragma unroll
        for (int r = 0; r < 8; ++r)
          t_f[wid][afrag_index(kc, r + 8 * grp, rel)] = (_Float16)fast_silu(c1[nt][r] + b1v[nt]);
      }
    }
    __syncthreads();

    // GEMM2: t[16x64] x W2[64x64]
    v8f c2[4];
#pragma unroll
    for (int nt = 0; nt < 4; ++nt) { v8f z = {}; c2[nt] = z; }
#pragma unroll
    for (int kc = 0; kc < 2; ++kc) {
      v16h a2 = frag_load(t_f[wid], kc);
#pragma unroll
      for (int nt = 0; nt < 4; ++nt) c2[nt] = WMMA_F16(a2, bw2[kc * 4 + nt], c2[nt]);
    }

    // gate -> gather h[src] -> atomic scatter-add to aggr[dst]
    if (active) {
#pragma unroll
      for (int r = 0; r < 8; ++r) {
        int mm = r + 8 * grp;
        long e = ebase + mm;
        if (e < (long)E) {
          const float* hp = h + (long)sidx[wid][mm] * HID + n;
          float* ap = aggr + (long)didx[wid][mm] * HID + n;
#pragma unroll
          for (int nt = 0; nt < 4; ++nt) {
            float g = fast_sigmoid(c2[nt][r] + b2v[nt]);
            atomic_add_f32_global(ap + nt * 16, g * hp[nt * 16]);
          }
        }
      }
    }
    __syncthreads();
  }
}

// ---------------------------------------------------------------------------
// Node update: h += MLP(concat(h, aggr)).  One wave per 16-node tile,
// 4 waves per block.  B fragments are single v16h LDS reads per use.
// ---------------------------------------------------------------------------
__global__ __launch_bounds__(128) void node_kernel(
    float* __restrict__ h, const float* __restrict__ aggr,
    const float* __restrict__ w1, const float* __restrict__ b1,
    const float* __restrict__ w2, const float* __restrict__ b2, int N) {
  __shared__ __align__(32) _Float16 w1h[16 * 512];      // 16 B-frags (K=128)
  __shared__ __align__(32) _Float16 w2h[8 * 512];       // 8 B-frags
  __shared__ __align__(32) _Float16 a_f[4][4 * 512];    // concat(h,aggr), A-frag layout
  __shared__ __align__(32) _Float16 t_f[4][2 * 512];    // SiLU intermediate, A-frag layout

  const int tid = threadIdx.x;
  const int wid = tid >> 5;
  const int lane = tid & 31;
  const int m = lane & 15;
  const int grp = lane >> 4;
  const int n = m;

  for (int i = tid; i < 128 * HID; i += 128) {
    int K = i >> 6, N2 = i & (HID - 1);
    w1h[bfrag_index(K, N2)] = (_Float16)w1[i];
  }
  for (int i = tid; i < 64 * HID; i += 128) {
    int K = i >> 6, N2 = i & (HID - 1);
    w2h[bfrag_index(K, N2)] = (_Float16)w2[i];
  }
  __syncthreads();

  float b1v[4], b2v[4];
#pragma unroll
  for (int nt = 0; nt < 4; ++nt) {
    b1v[nt] = b1[nt * 16 + n];
    b2v[nt] = b2[nt * 16 + n];
  }

  const int ntiles = (N + 15) >> 4;
  const int waves_total = gridDim.x * 4;
  const int iters = (ntiles + waves_total - 1) / waves_total;

  for (int it = 0; it < iters; ++it) {
    const int tile = it * waves_total + blockIdx.x * 4 + wid;
    const bool active = (tile < ntiles);
    const int nbase = tile << 4;

    // Stage concat(h, aggr) as f16 [16 x 128] in A-frag layout.
    if (active) {
      int node = nbase + m;
      for (int j = 0; j < 64; ++j) {
        int col = (grp << 6) + j;
        float v = 0.f;
        if (node < N)
          v = (col < HID) ? h[(long)node * HID + col] : aggr[(long)node * HID + col - HID];
        a_f[wid][afrag_index(col >> 5, m, col & 31)] = (_Float16)v;
      }
    }
    __syncthreads();

    // GEMM1: [16x128] x [128x64]
    v8f c1[4];
#pragma unroll
    for (int nt = 0; nt < 4; ++nt) { v8f z = {}; c1[nt] = z; }
#pragma unroll
    for (int kc = 0; kc < 4; ++kc) {
      v16h aa = frag_load(a_f[wid], kc);
#pragma unroll
      for (int nt = 0; nt < 4; ++nt)
        c1[nt] = WMMA_F16(aa, frag_load(w1h, kc * 4 + nt), c1[nt]);
    }
    if (active) {
#pragma unroll
      for (int nt = 0; nt < 4; ++nt) {
        int kc = nt >> 1;
        int rel = ((nt & 1) << 4) + n;
#pragma unroll
        for (int r = 0; r < 8; ++r)
          t_f[wid][afrag_index(kc, r + 8 * grp, rel)] = (_Float16)fast_silu(c1[nt][r] + b1v[nt]);
      }
    }
    __syncthreads();

    // GEMM2: [16x64] x [64x64], residual add into h.
    v8f c2[4];
#pragma unroll
    for (int nt = 0; nt < 4; ++nt) { v8f z = {}; c2[nt] = z; }
#pragma unroll
    for (int kc = 0; kc < 2; ++kc) {
      v16h aa = frag_load(t_f[wid], kc);
#pragma unroll
      for (int nt = 0; nt < 4; ++nt)
        c2[nt] = WMMA_F16(aa, frag_load(w2h, kc * 4 + nt), c2[nt]);
    }
    if (active) {
#pragma unroll
      for (int r = 0; r < 8; ++r) {
        int node = nbase + r + 8 * grp;
        if (node < N) {
          float* hp = h + (long)node * HID + n;
#pragma unroll
          for (int nt = 0; nt < 4; ++nt) hp[nt * 16] += c2[nt][r] + b2v[nt];
        }
      }
    }
    __syncthreads();
  }
}

// ---------------------------------------------------------------------------
// Final LayerNorm (in place): one wave per node row of 64.
// ---------------------------------------------------------------------------
__global__ __launch_bounds__(256) void ln_kernel(float* __restrict__ h,
                                                 const float* __restrict__ g,
                                                 const float* __restrict__ b, int N) {
  int row = blockIdx.x * 8 + (threadIdx.x >> 5);
  int lane = threadIdx.x & 31;
  if (row >= N) return;
  float x0 = h[(long)row * HID + lane];
  float x1 = h[(long)row * HID + 32 + lane];
  float s = x0 + x1;
#pragma unroll
  for (int off = 16; off > 0; off >>= 1) s += __shfl_xor(s, off, 32);
  float mu = s * (1.f / 64.f);
  float d0 = x0 - mu, d1 = x1 - mu;
  float v = d0 * d0 + d1 * d1;
#pragma unroll
  for (int off = 16; off > 0; off >>= 1) v += __shfl_xor(v, off, 32);
  float inv = __builtin_amdgcn_rsqf(v * (1.f / 64.f) + 1e-5f);
  h[(long)row * HID + lane]      = d0 * inv * g[lane] + b[lane];
  h[(long)row * HID + 32 + lane] = d1 * inv * g[lane + 32] + b[lane + 32];
}

// ---------------------------------------------------------------------------
extern "C" void kernel_launch(void* const* d_in, const int* in_sizes, int n_in,
                              void* d_out, int out_size, void* d_ws, size_t ws_size,
                              hipStream_t stream) {
  const float* charges = (const float*)d_in[0];
  const float* dist    = (const float*)d_in[1];
  const int*   eidx    = (const int*)d_in[2];
  const float* embed_w = (const float*)d_in[3];
  const float* embed_b = (const float*)d_in[4];
  const float* ew1     = (const float*)d_in[5];
  const float* eb1     = (const float*)d_in[6];
  const float* ew2     = (const float*)d_in[7];
  const float* eb2     = (const float*)d_in[8];
  const float* uw1     = (const float*)d_in[9];
  const float* ub1     = (const float*)d_in[10];
  const float* uw2     = (const float*)d_in[11];
  const float* ub2     = (const float*)d_in[12];
  const float* lng     = (const float*)d_in[13];
  const float* lnb     = (const float*)d_in[14];

  const int N = in_sizes[0];
  const int E = in_sizes[1];
  const int* src = eidx;
  const int* dst = eidx + E;

  float* h    = (float*)d_out;  // h lives in the output buffer (fully rewritten each call)
  float* aggr = (float*)d_ws;   // N*HID floats of scratch

  int total = N * HID;
  embed_kernel<<<(total + 255) / 256, 256, 0, stream>>>(charges, embed_w, embed_b, h, total);

  for (int l = 0; l < 3; ++l) {
    hipMemsetAsync(aggr, 0, (size_t)N * HID * sizeof(float), stream);
    edge_kernel<<<1024, 256, 0, stream>>>(dist, src, dst,
                                          ew1 + (size_t)l * NRBF * HID, eb1 + (size_t)l * HID,
                                          ew2 + (size_t)l * HID * HID, eb2 + (size_t)l * HID,
                                          h, aggr, E);
    node_kernel<<<256, 128, 0, stream>>>(h, aggr,
                                         uw1 + (size_t)l * 2 * HID * HID, ub1 + (size_t)l * HID,
                                         uw2 + (size_t)l * HID * HID, ub2 + (size_t)l * HID, N);
  }
  ln_kernel<<<(N + 7) / 8, 256, 0, stream>>>(h, lng, lnb, N);
}
